// Grid4DEncoder_68092411511050
// MI455X (gfx1250) — compile-verified, matrix operand order checked by reading, not run on hardware
//
#include <hip/hip_runtime.h>
#include <hip/hip_bf16.h>
#include <stdint.h>

// ---------------- problem constants ----------------
#define BATCH        262144
#define NS_LEVELS    16
#define NT_LEVELS    8
#define F_PER_LEVEL  2
#define S_TABLE      524288
#define T_TABLE      131072
#define HIDDEN       512
#define KPAD         64          // 48 features padded to 64 for K%32==0
#define HPAD         520         // hBuf LDS row stride (halfs): 260 dwords = 4 mod 64 -> conflict-free b128
#define ROWS_PER_BLK 64          // M rows per block (4 WMMA M-tiles per wave)

typedef __attribute__((ext_vector_type(16))) _Float16 v16h;
typedef __attribute__((ext_vector_type(8)))  float    v8f;

__constant__ float kSpatialRes[NS_LEVELS] = {
    16.f, 22.f, 30.f, 42.f, 58.f, 80.f, 111.f, 154.f,
    212.f, 294.f, 406.f, 561.f, 776.f, 1072.f, 1482.f, 2048.f };
__constant__ float kTemporalRes[NT_LEVELS] = {
    8.f, 16.f, 32.f, 64.f, 128.f, 256.f, 512.f, 1024.f };

// ---------------- weight prep: fp32 [K][N] -> f16 [N][K] ----------------
__global__ void grid4d_prep_w1(const float* __restrict__ w1, _Float16* __restrict__ w1t) {
    int id = blockIdx.x * 256 + threadIdx.x;      // 512*64 threads
    int n = id >> 6;
    int k = id & 63;
    w1t[id] = (k < (NS_LEVELS + NT_LEVELS) * F_PER_LEVEL)
                  ? (_Float16)w1[k * HIDDEN + n]
                  : (_Float16)0.0f;
}

__global__ void grid4d_prep_w2(const float* __restrict__ w2, _Float16* __restrict__ w2t) {
    int id = blockIdx.x * 256 + threadIdx.x;      // 512*512 threads
    int n = id >> 9;
    int k = id & 511;
    w2t[id] = (_Float16)w2[k * HIDDEN + n];
}

// ---------------- hash-grid encode: one thread per point ----------------
__global__ void grid4d_encode(const float* __restrict__ xyzt,
                              const unsigned char* __restrict__ smask,
                              const unsigned char* __restrict__ tmask,
                              const float* __restrict__ stab,
                              const float* __restrict__ ttab,
                              _Float16* __restrict__ feat) {
    int b = blockIdx.x * 256 + threadIdx.x;
    if (b >= BATCH) return;

    const float x = xyzt[b * 4 + 0];
    const float y = xyzt[b * 4 + 1];
    const float z = xyzt[b * 4 + 2];
    const float t = xyzt[b * 4 + 3];
    const float sm = smask[b] ? 1.0f : 0.0f;
    const float tm = tmask[b] ? 1.0f : 0.0f;

    _Float16 out[KPAD];

    // spatial levels: 3D trilinear, hash = x*1 ^ y*2654435761 ^ z*805459861
    #pragma unroll
    for (int l = 0; l < NS_LEVELS; ++l) {
        const float res = kSpatialRes[l];
        const float px = x * res, py = y * res, pz = z * res;
        const float fx = floorf(px), fy = floorf(py), fz = floorf(pz);
        const float rx = px - fx, ry = py - fy, rz = pz - fz;
        const uint32_t bx = (uint32_t)fx, by = (uint32_t)fy, bz = (uint32_t)fz;
        const float* __restrict__ tab = stab + (size_t)l * (S_TABLE * F_PER_LEVEL);
        float f0 = 0.0f, f1 = 0.0f;
        #pragma unroll
        for (int c = 0; c < 8; ++c) {
            const uint32_t ox = (c >> 2) & 1, oy = (c >> 1) & 1, oz = c & 1;
            uint32_t h = (bx + ox) * 1u;
            h ^= (by + oy) * 2654435761u;
            h ^= (bz + oz) * 805459861u;
            const uint32_t idx = h & (uint32_t)(S_TABLE - 1);
            const float2 f = *(const float2*)(tab + (size_t)idx * 2);
            const float w = (ox ? rx : 1.0f - rx) *
                            (oy ? ry : 1.0f - ry) *
                            (oz ? rz : 1.0f - rz);
            f0 = fmaf(w, f.x, f0);
            f1 = fmaf(w, f.y, f1);
        }
        out[2 * l + 0] = (_Float16)(f0 * sm);
        out[2 * l + 1] = (_Float16)(f1 * sm);
    }

    // temporal levels: 1D linear, hash = x*1
    #pragma unroll
    for (int l = 0; l < NT_LEVELS; ++l) {
        const float res = kTemporalRes[l];
        const float pt = t * res;
        const float ft = floorf(pt);
        const float rt = pt - ft;
        const uint32_t bt = (uint32_t)ft;
        const float* __restrict__ tab = ttab + (size_t)l * (T_TABLE * F_PER_LEVEL);
        const uint32_t i0 = bt & (uint32_t)(T_TABLE - 1);
        const uint32_t i1 = (bt + 1u) & (uint32_t)(T_TABLE - 1);
        const float2 fa = *(const float2*)(tab + (size_t)i0 * 2);
        const float2 fb = *(const float2*)(tab + (size_t)i1 * 2);
        const float f0 = (1.0f - rt) * fa.x + rt * fb.x;
        const float f1 = (1.0f - rt) * fa.y + rt * fb.y;
        out[2 * NS_LEVELS + 2 * l + 0] = (_Float16)(f0 * tm);
        out[2 * NS_LEVELS + 2 * l + 1] = (_Float16)(f1 * tm);
    }

    #pragma unroll
    for (int k = (NS_LEVELS + NT_LEVELS) * F_PER_LEVEL; k < KPAD; ++k)
        out[k] = (_Float16)0.0f;

    // 128B coalesced store (one full cacheline per point)
    float4* dst = (float4*)(feat + (size_t)b * KPAD);
    const float4* src = (const float4*)out;
    #pragma unroll
    for (int i = 0; i < 8; ++i) dst[i] = src[i];
}

// ---------------- fused GEMM1 + bias + LN + GELU + GEMM2 + bias ----------
// block = 256 threads = 8 wave32; each block does 64 rows of the batch.
// wave w owns output columns [w*64, w*64+64) (4 N-tiles) for ALL 64 rows
// (4 M-tiles) -> every B fragment is reused by 4 WMMAs, quartering the
// L2 traffic for w2t vs an M=16 tiling.
__global__ __launch_bounds__(256) void grid4d_mlp(
    const _Float16* __restrict__ feat,  // [B][64]    f16
    const _Float16* __restrict__ w1t,   // [512][64]  f16, N-major
    const float*    __restrict__ b1,
    const float*    __restrict__ ln_g,
    const float*    __restrict__ ln_b,
    const _Float16* __restrict__ w2t,   // [512][512] f16, N-major
    const float*    __restrict__ b2,
    float*          __restrict__ out) {

    __shared__ _Float16 hBuf[ROWS_PER_BLK * HPAD];   // ~65 KB, only LDS buffer

    const int tid   = threadIdx.x;
    const int wave  = tid >> 5;
    const int lane  = tid & 31;
    const int rowbase = blockIdx.x * ROWS_PER_BLK;

    const int m16   = lane & 15;          // M (A frag) / N-within-tile (B frag)
    const int khalf = (lane >> 4) * 8;    // per-ISA half-wave K offset
    const int mrow0 = (lane >> 4) * 8;    // C/D layout row offset
    const int nbase = wave * 64;

    // ================= GEMM1: [64 x 64] @ [64 x 512] =================
    v8f acc[4][4];
    #pragma unroll
    for (int mt = 0; mt < 4; ++mt)
        #pragma unroll
        for (int tI = 0; tI < 4; ++tI) acc[mt][tI] = (v8f){};

    #pragma unroll
    for (int kk = 0; kk < 2; ++kk) {
        const int kb = kk * 32 + khalf;
        v16h a[4];
        #pragma unroll
        for (int mt = 0; mt < 4; ++mt) {
            const _Float16* ap =
                feat + (size_t)(rowbase + mt * 16 + m16) * KPAD + kb;
            ((float4*)&a[mt])[0] = *(const float4*)(ap);
            ((float4*)&a[mt])[1] = *(const float4*)(ap + 16);
        }
        #pragma unroll
        for (int tI = 0; tI < 4; ++tI) {
            const int n = nbase + tI * 16 + m16;
            v16h bf;
            ((float4*)&bf)[0] = *(const float4*)&w1t[n * KPAD + kb];
            ((float4*)&bf)[1] = *(const float4*)&w1t[n * KPAD + kb + 16];
            #pragma unroll
            for (int mt = 0; mt < 4; ++mt)
                acc[mt][tI] = __builtin_amdgcn_wmma_f32_16x16x32_f16(
                    false, a[mt], false, bf, (short)0, acc[mt][tI], false, false);
        }
    }

    // spill (+b1) as f16 into LDS using the documented C/D layout:
    // VGPR r -> M = mt*16 + r + 8*(lane>>4), N = (lane&15) + tile*16
    #pragma unroll
    for (int tI = 0; tI < 4; ++tI) {
        const int n = nbase + tI * 16 + m16;
        const float bias = b1[n];
        #pragma unroll
        for (int mt = 0; mt < 4; ++mt)
            #pragma unroll
            for (int r = 0; r < 8; ++r)
                hBuf[(mt * 16 + mrow0 + r) * HPAD + n] =
                    (_Float16)(acc[mt][tI][r] + bias);
    }
    __syncthreads();

    // ============ LayerNorm + exact GELU, in place, 8 rows per wave =====
    {
        float g[16], bb[16];
        #pragma unroll
        for (int j = 0; j < 16; ++j) {
            g[j]  = ln_g[lane + 32 * j];
            bb[j] = ln_b[lane + 32 * j];
        }
        #pragma unroll
        for (int rr = 0; rr < 8; ++rr) {
            const int row = wave * 8 + rr;
            float v[16];
            float s = 0.0f, s2 = 0.0f;
            #pragma unroll
            for (int j = 0; j < 16; ++j) {
                const float xv = (float)hBuf[row * HPAD + lane + 32 * j];
                v[j] = xv;
                s += xv;
                s2 = fmaf(xv, xv, s2);
            }
            #pragma unroll
            for (int off = 16; off > 0; off >>= 1) {
                s  += __shfl_xor(s,  off, 32);
                s2 += __shfl_xor(s2, off, 32);
            }
            const float mu   = s * (1.0f / 512.0f);
            const float var  = s2 * (1.0f / 512.0f) - mu * mu;
            const float rsig = rsqrtf(var + 1e-5f);
            #pragma unroll
            for (int j = 0; j < 16; ++j) {
                const float u  = (v[j] - mu) * rsig * g[j] + bb[j];
                const float ge = 0.5f * u * (1.0f + erff(u * 0.70710678118654752f));
                hBuf[row * HPAD + lane + 32 * j] = (_Float16)ge;
            }
        }
    }
    __syncthreads();

    // ================= GEMM2: [64 x 512] @ [512 x 512] =================
    v8f acc2[4][4];
    #pragma unroll
    for (int mt = 0; mt < 4; ++mt)
        #pragma unroll
        for (int tI = 0; tI < 4; ++tI) acc2[mt][tI] = (v8f){};

    #pragma unroll
    for (int kk = 0; kk < 16; ++kk) {
        const int kb = kk * 32 + khalf;
        v16h a[4];
        #pragma unroll
        for (int mt = 0; mt < 4; ++mt) {
            const _Float16* ap = &hBuf[(mt * 16 + m16) * HPAD + kb];
            ((float4*)&a[mt])[0] = *(const float4*)(ap);
            ((float4*)&a[mt])[1] = *(const float4*)(ap + 16);
        }
        #pragma unroll
        for (int tI = 0; tI < 4; ++tI) {
            const int n = nbase + tI * 16 + m16;
            v16h bf;
            ((float4*)&bf)[0] = *(const float4*)&w2t[(size_t)n * HIDDEN + kb];
            ((float4*)&bf)[1] = *(const float4*)&w2t[(size_t)n * HIDDEN + kb + 16];
            #pragma unroll
            for (int mt = 0; mt < 4; ++mt)
                acc2[mt][tI] = __builtin_amdgcn_wmma_f32_16x16x32_f16(
                    false, a[mt], false, bf, (short)0, acc2[mt][tI], false, false);
        }
    }

    // ---- +b2, store f32 output ----
    #pragma unroll
    for (int tI = 0; tI < 4; ++tI) {
        const int n = nbase + tI * 16 + m16;
        const float bias = b2[n];
        #pragma unroll
        for (int mt = 0; mt < 4; ++mt)
            #pragma unroll
            for (int r = 0; r < 8; ++r)
                out[(size_t)(rowbase + mt * 16 + mrow0 + r) * HIDDEN + n] =
                    acc2[mt][tI][r] + bias;
    }
}

// ---------------- host launcher ----------------
extern "C" void kernel_launch(void* const* d_in, const int* in_sizes, int n_in,
                              void* d_out, int out_size, void* d_ws, size_t ws_size,
                              hipStream_t stream) {
    (void)in_sizes; (void)n_in; (void)out_size; (void)ws_size;

    const float*         xyzt  = (const float*)d_in[0];
    const unsigned char* smask = (const unsigned char*)d_in[1];
    const unsigned char* tmask = (const unsigned char*)d_in[2];
    const float*         stab  = (const float*)d_in[3];
    const float*         ttab  = (const float*)d_in[4];
    const float*         w1    = (const float*)d_in[5];
    const float*         b1    = (const float*)d_in[6];
    const float*         ln_g  = (const float*)d_in[7];
    const float*         ln_b  = (const float*)d_in[8];
    const float*         w2    = (const float*)d_in[9];
    const float*         b2    = (const float*)d_in[10];
    float*               out   = (float*)d_out;

    char* ws = (char*)d_ws;
    _Float16* w1t  = (_Float16*)(ws);                                 //  64 KB
    _Float16* w2t  = (_Float16*)(ws + 65536);                         // 512 KB
    _Float16* feat = (_Float16*)(ws + 65536 + 524288);                //  32 MB

    grid4d_prep_w1<<<(HIDDEN * KPAD) / 256, 256, 0, stream>>>(w1, w1t);
    grid4d_prep_w2<<<(HIDDEN * HIDDEN) / 256, 256, 0, stream>>>(w2, w2t);
    grid4d_encode<<<BATCH / 256, 256, 0, stream>>>(xyzt, smask, tmask, stab, ttab, feat);
    grid4d_mlp<<<BATCH / ROWS_PER_BLK, 256, 0, stream>>>(feat, w1t, b1, ln_g, ln_b, w2t, b2, out);
}
